// LSTM_78589311582789
// MI455X (gfx1250) — matrix operand order repaired and checked by reading, not compiled, not run on hardware
//
#include <hip/hip_runtime.h>

// Problem constants (match reference).
#define BB 64
#define TT 2048
#define INDIM 12
#define HH 128
#define NCC 256
#define BT (BB * TT)

typedef __attribute__((ext_vector_type(16))) __bf16 v16bf;
typedef __attribute__((ext_vector_type(16))) unsigned short v16us;
typedef __attribute__((ext_vector_type(4))) unsigned short v4us;
typedef __attribute__((ext_vector_type(8))) float v8f;
typedef __attribute__((ext_vector_type(4))) float v4f;

__device__ __forceinline__ unsigned short f2bf(float f) {
    unsigned u = __builtin_bit_cast(unsigned, f);
    u += 0x7FFFu + ((u >> 16) & 1u);   // round-to-nearest-even
    return (unsigned short)(u >> 16);
}
__device__ __forceinline__ float bf2f(unsigned short s) {
    return __builtin_bit_cast(float, ((unsigned)s) << 16);
}
__device__ __forceinline__ v8f wmma_bf16f32(v16us a, v16us b, v8f c) {
    return __builtin_amdgcn_wmma_f32_16x16x32_bf16(
        false, __builtin_bit_cast(v16bf, a),
        false, __builtin_bit_cast(v16bf, b),
        (short)0, c, false, false);
}

// Branch-free activations: v_exp_f32 + v_rcp_f32, no divergence on the
// sequential scan critical path (libm tanhf has an |x| range-reduction branch).
__device__ __forceinline__ float fast_sigm(float x) {
#if __has_builtin(__builtin_amdgcn_exp2f) && __has_builtin(__builtin_amdgcn_rcpf)
    float e = __builtin_amdgcn_exp2f(x * -1.442695040888963f);
    return __builtin_amdgcn_rcpf(1.0f + e);
#else
    return 1.0f / (1.0f + __expf(-x));
#endif
}
__device__ __forceinline__ float fast_tanh(float x) {
#if __has_builtin(__builtin_amdgcn_exp2f) && __has_builtin(__builtin_amdgcn_rcpf)
    float e = __builtin_amdgcn_exp2f(x * -2.885390081777927f);  // exp2(-2x*log2e)
    return 2.0f * __builtin_amdgcn_rcpf(1.0f + e) - 1.0f;
#else
    return tanhf(x);
#endif
}

// ---------------------------------------------------------------------------
// Small conversion kernels (run once per call, negligible cost)
// ---------------------------------------------------------------------------
__global__ void cvt_bf16_kernel(const float* __restrict__ s, unsigned short* __restrict__ d, int n) {
    int i = blockIdx.x * 256 + threadIdx.x;
    if (i < n) d[i] = f2bf(s[i]);
}
// src R x C (row-major) -> dst C x R (row-major) bf16  (transpose + convert)
__global__ void cvt_bf16_tr_kernel(const float* __restrict__ s, unsigned short* __restrict__ d, int R, int C) {
    int i = blockIdx.x * 256 + threadIdx.x;
    if (i < R * C) {
        int r = i / C, c = i % C;
        d[(size_t)c * R + r] = f2bf(s[i]);
    }
}
__global__ void bias_sum_kernel(const float* __restrict__ a, const float* __restrict__ b,
                                float* __restrict__ o, int n) {
    int i = blockIdx.x * 256 + threadIdx.x;
    if (i < n) o[i] = a[i] + b[i];
}

// ---------------------------------------------------------------------------
// Generic WMMA GEMM:  Out(MxN,f32 or bf16) = A(MxK,bf16) * B(KxN,bf16) + bias(N)
// Block = 256 threads (8 waves). Block tile 128x128; wave = 16 rows x 128 cols.
// ---------------------------------------------------------------------------
template <bool OUT_BF16>
__global__ void __launch_bounds__(256)
gemm_bias_kernel(const unsigned short* __restrict__ A,
                 const unsigned short* __restrict__ Bm,
                 const float* __restrict__ bias,
                 void* __restrict__ Out,
                 int M, int K, int N) {
    __shared__ alignas(16) unsigned short aT[128][34];   // 128 rows x 32 k (padded)
    __shared__ alignas(16) unsigned short bT[32][136];   // 32 k x 128 n (padded)

    const int tid = threadIdx.x, wave = tid >> 5, lane = tid & 31;
    const int m0 = blockIdx.x << 7, n0 = blockIdx.y << 7;

    v8f acc[8];
#pragma unroll
    for (int i = 0; i < 8; ++i) acc[i] = {};

    for (int kb = 0; kb < K; kb += 32) {
        __syncthreads();
        // Stage A tile (each thread: 16 contiguous k of one row-half)
#pragma unroll
        for (int e = 0; e < 16; ++e) {
            int idx = tid * 16 + e;
            int r = idx >> 5, k = idx & 31;
            aT[r][k] = (kb + k < K) ? A[(size_t)(m0 + r) * K + kb + k] : (unsigned short)0;
        }
        // Stage B tile (coalesced along n)
#pragma unroll
        for (int e = 0; e < 16; ++e) {
            int idx = tid + (e << 8);
            int r = idx >> 7, n = idx & 127;
            bT[r][n] = (kb + r < K) ? Bm[(size_t)(kb + r) * N + n0 + n] : (unsigned short)0;
        }
        __syncthreads();

        // A fragment: 16x32, M = lane&15 (+wave*16 rows), k per ISA layout
        v16us au;
#pragma unroll
        for (int j = 0; j < 16; ++j) {
            int k = j + ((j >> 3) << 3) + ((lane >> 4) << 3);
            au[j] = aT[(wave << 4) + (lane & 15)][k];
        }
#pragma unroll
        for (int nt = 0; nt < 8; ++nt) {
            v16us bu;
#pragma unroll
            for (int j = 0; j < 16; ++j) {
                int k = j + ((j >> 3) << 3) + ((lane >> 4) << 3);
                bu[j] = bT[k][(nt << 4) + (lane & 15)];
            }
            acc[nt] = wmma_bf16f32(au, bu, acc[nt]);
        }
    }

    // Epilogue: C/D layout M = r + 8*(lane>=16), N = lane&15
#pragma unroll
    for (int nt = 0; nt < 8; ++nt) {
        int n = n0 + (nt << 4) + (lane & 15);
        float bv = bias[n];
#pragma unroll
        for (int r = 0; r < 8; ++r) {
            int m = m0 + (wave << 4) + r + ((lane >> 4) << 3);
            float v = acc[nt][r] + bv;
            if constexpr (OUT_BF16)
                ((unsigned short*)Out)[(size_t)m * N + n] = f2bf(v);
            else
                ((float*)Out)[(size_t)m * N + n] = v;
        }
    }
}

// ---------------------------------------------------------------------------
// LSTM recurrence. One block per (direction, batch-tile of 16 rows).
// 512 threads = 16 waves; wave w owns gate columns [32w, 32w+32).
// WhhT fragments live in VGPRs for the entire T loop; h (bf16) and c (f32)
// live in LDS. Per step: prefetch pre (b64 x4) -> 8 WMMAs/wave -> gate buffer
// -> branch-free fused gate math with vectorized (b128/b64) accesses.
// ---------------------------------------------------------------------------
__global__ void __launch_bounds__(512)
lstm_scan_kernel(const unsigned short* __restrict__ pre,   // [2][B*T][512] bf16
                 size_t preStride,                          // elements per direction
                 const unsigned short* __restrict__ whhT_f, // 128x512 bf16
                 const unsigned short* __restrict__ whhT_b, // 128x512 bf16
                 unsigned short* __restrict__ actOut,       // [B*T][256] bf16 (fwd|bwd)
                 int T) {
    constexpr int GLD = 520;  // padded row stride of gate buffer (floats)
    constexpr int HLD = 136;  // padded row stride of h buffer (ushorts)
    __shared__ alignas(16) float gbuf[16 * GLD];
    __shared__ alignas(16) float cbuf[16 * 128];
    __shared__ alignas(16) unsigned short hbuf[16 * HLD];

    const int dir = blockIdx.x >> 2;
    const int bt = blockIdx.x & 3;
    const unsigned short* __restrict__ whhT = dir ? whhT_b : whhT_f;
    const unsigned short* __restrict__ preD = pre + (size_t)dir * preStride;

    const int tid = threadIdx.x, wave = tid >> 5, lane = tid & 31;

    for (int i = tid; i < 16 * 128; i += 512) cbuf[i] = 0.f;
    for (int i = tid; i < 16 * HLD; i += 512) hbuf[i] = 0;
    __syncthreads();

    // Preload loop-invariant Whh^T fragments into registers (64 VGPRs/wave).
    v16us bfr[2][4];
#pragma unroll
    for (int nt = 0; nt < 2; ++nt)
#pragma unroll
        for (int kc = 0; kc < 4; ++kc)
#pragma unroll
            for (int j = 0; j < 16; ++j) {
                int n = (wave << 5) + (nt << 4) + (lane & 15);
                int k = (kc << 5) + j + ((j >> 3) << 3) + ((lane >> 4) << 3);
                bfr[nt][kc][j] = whhT[(size_t)k * 512 + n];
            }

    // Elementwise ownership: 4 consecutive h-columns of one row per thread.
    const int erow = tid >> 5;              // 0..15
    const int ehc = (tid & 31) << 2;        // 0,4,...,124
    const int eb = (bt << 4) + erow;        // global batch row

    for (int s = 0; s < T; ++s) {
        const int t = dir ? (T - 1 - s) : s;

        // Prefetch this step's pre-activations (independent of h_prev) so the
        // global loads overlap the recurrent WMMA work.
        const size_t pbase = ((size_t)eb * T + t) << 9;  // *512
        v4us pi = *(const v4us*)(preD + pbase + ehc);
        v4us pf = *(const v4us*)(preD + pbase + 128 + ehc);
        v4us pg = *(const v4us*)(preD + pbase + 256 + ehc);
        v4us po = *(const v4us*)(preD + pbase + 384 + ehc);

        // g = h_prev @ WhhT  (16x512, K=128 in 4 chunks of 32)
        v8f acc0 = {}, acc1 = {};
#pragma unroll
        for (int kc = 0; kc < 4; ++kc) {
            v16us au;
#pragma unroll
            for (int j = 0; j < 16; ++j) {
                int k = (kc << 5) + j + ((j >> 3) << 3) + ((lane >> 4) << 3);
                au[j] = hbuf[(lane & 15) * HLD + k];
            }
            acc0 = wmma_bf16f32(au, bfr[0][kc], acc0);
            acc1 = wmma_bf16f32(au, bfr[1][kc], acc1);
        }
#pragma unroll
        for (int r = 0; r < 8; ++r) {
            int row = r + ((lane >> 4) << 3);
            gbuf[row * GLD + (wave << 5) + (lane & 15)] = acc0[r];
            gbuf[row * GLD + (wave << 5) + 16 + (lane & 15)] = acc1[r];
        }
        __syncthreads();

        // Gate math (branch-free), vectorized 4-wide. Gate order i,f,g,o.
        v4f gi = *(const v4f*)(gbuf + erow * GLD + ehc);
        v4f gf = *(const v4f*)(gbuf + erow * GLD + 128 + ehc);
        v4f gg = *(const v4f*)(gbuf + erow * GLD + 256 + ehc);
        v4f go = *(const v4f*)(gbuf + erow * GLD + 384 + ehc);
        v4f cc = *(const v4f*)(cbuf + erow * 128 + ehc);
        v4us hb;
#pragma unroll
        for (int u = 0; u < 4; ++u) {
            float iv = gi[u] + bf2f(pi[u]);
            float fv = gf[u] + bf2f(pf[u]);
            float gv = gg[u] + bf2f(pg[u]);
            float ov = go[u] + bf2f(po[u]);
            float c = fast_sigm(fv) * cc[u] + fast_sigm(iv) * fast_tanh(gv);
            float h = fast_sigm(ov) * fast_tanh(c);
            cc[u] = c;
            hb[u] = f2bf(h);
        }
        *(v4f*)(cbuf + erow * 128 + ehc) = cc;
        *(v4us*)(hbuf + erow * HLD + ehc) = hb;
        *(v4us*)(actOut + (((size_t)eb * T + t) << 8) + (dir << 7) + ehc) = hb;
        __syncthreads();
    }
}

// ---------------------------------------------------------------------------
// Host orchestration
// ---------------------------------------------------------------------------
extern "C" void kernel_launch(void* const* d_in, const int* in_sizes, int n_in,
                              void* d_out, int out_size, void* d_ws, size_t ws_size,
                              hipStream_t stream) {
    (void)in_sizes; (void)n_in; (void)out_size; (void)ws_size;

    const float* x = (const float*)d_in[0];
    const float* Wih[2][2]; const float* Whh[2][2];
    const float* bih[2][2]; const float* bhh[2][2];
    int p = 1;
    for (int l = 0; l < 2; ++l)
        for (int d = 0; d < 2; ++d) {
            Wih[l][d] = (const float*)d_in[p++];
            Whh[l][d] = (const float*)d_in[p++];
            bih[l][d] = (const float*)d_in[p++];
            bhh[l][d] = (const float*)d_in[p++];
        }
    const float* fc_w = (const float*)d_in[p++];
    const float* fc_b = (const float*)d_in[p++];

    char* base = (char*)d_ws;
    size_t off = 0;
    auto alloc = [&](size_t bytes) -> void* {
        void* r = base + off;
        off = (off + bytes + 255) & ~(size_t)255;
        return r;
    };

    unsigned short* xb = (unsigned short*)alloc((size_t)BT * INDIM * 2);
    unsigned short* wihT[2][2]; unsigned short* whhT[2][2]; float* biasS[2][2];
    for (int l = 0; l < 2; ++l)
        for (int d = 0; d < 2; ++d) {
            int indim = (l == 0) ? INDIM : 2 * HH;
            wihT[l][d] = (unsigned short*)alloc((size_t)indim * 4 * HH * 2);
            whhT[l][d] = (unsigned short*)alloc((size_t)HH * 4 * HH * 2);
            biasS[l][d] = (float*)alloc((size_t)4 * HH * 4);
        }
    unsigned short* fcwT = (unsigned short*)alloc((size_t)NCC * 2 * HH * 2);
    unsigned short* actA = (unsigned short*)alloc((size_t)BT * 2 * HH * 2);
    unsigned short* actB = (unsigned short*)alloc((size_t)BT * 2 * HH * 2);
    const size_t preElems = (size_t)BT * 4 * HH;
    unsigned short* pre = (unsigned short*)alloc(preElems * 2 * 2);  // both dirs, bf16

    // One-time conversions (f32 -> bf16, weights transposed to K x N).
    {
        int n = BT * INDIM;
        cvt_bf16_kernel<<<(n + 255) / 256, 256, 0, stream>>>(x, xb, n);
    }
    for (int l = 0; l < 2; ++l)
        for (int d = 0; d < 2; ++d) {
            int indim = (l == 0) ? INDIM : 2 * HH;
            int n1 = 4 * HH * indim;
            cvt_bf16_tr_kernel<<<(n1 + 255) / 256, 256, 0, stream>>>(Wih[l][d], wihT[l][d], 4 * HH, indim);
            int n2 = 4 * HH * HH;
            cvt_bf16_tr_kernel<<<(n2 + 255) / 256, 256, 0, stream>>>(Whh[l][d], whhT[l][d], 4 * HH, HH);
            bias_sum_kernel<<<2, 256, 0, stream>>>(bih[l][d], bhh[l][d], biasS[l][d], 4 * HH);
        }
    {
        int n = NCC * 2 * HH;
        cvt_bf16_tr_kernel<<<(n + 255) / 256, 256, 0, stream>>>(fc_w, fcwT, NCC, 2 * HH);
    }

    dim3 gPre(BT / 128, (4 * HH) / 128);  // 1024 x 4

    // Layer 0: input projection (K=12) then scan.
    for (int d = 0; d < 2; ++d)
        gemm_bias_kernel<true><<<gPre, 256, 0, stream>>>(
            xb, wihT[0][d], biasS[0][d], pre + (size_t)d * preElems, BT, INDIM, 4 * HH);
    lstm_scan_kernel<<<8, 512, 0, stream>>>(pre, preElems, whhT[0][0], whhT[0][1], actA, TT);

    // Layer 1: input projection (K=256) then scan.
    for (int d = 0; d < 2; ++d)
        gemm_bias_kernel<true><<<gPre, 256, 0, stream>>>(
            actA, wihT[1][d], biasS[1][d], pre + (size_t)d * preElems, BT, 2 * HH, 4 * HH);
    lstm_scan_kernel<<<8, 512, 0, stream>>>(pre, preElems, whhT[1][0], whhT[1][1], actB, TT);

    // Final FC into d_out (f32).
    dim3 gFc(BT / 128, NCC / 128);
    gemm_bias_kernel<false><<<gFc, 256, 0, stream>>>(actB, fcwT, fc_b, (float*)d_out, BT, 2 * HH, NCC);
}
